// PINN_52123723104357
// MI455X (gfx1250) — matrix operand order repaired and checked by reading, not compiled, not run on hardware
//
#include <hip/hip_runtime.h>

typedef __attribute__((ext_vector_type(16))) _Float16 v16h;
typedef __attribute__((ext_vector_type(8)))  _Float16 v8h;
typedef __attribute__((ext_vector_type(8)))  float    v8f;

union HV { v16h v16; v8h v8[2]; };

#define XSTR 72          // act row stride in halves (bank-conflict friendly)
#define NFRAG 42         // 7 layers * 2 K-chunks * 3 N-tiles
#define FRAG_HALVES (NFRAG * 32 * 16)   // 21504 halves = 43008 bytes
#define BIAS_FLOATS (7 * 48)            // 336 floats
#define ACT_ROWS 32                      // two 16-row sub-tiles per wave
#define ACT_BUF_HALVES (ACT_ROWS * XSTR)
#define ACT_HALVES (8 * 2 * ACT_BUF_HALVES)

__device__ __forceinline__ float tanh_fast(float x) {
    // tanh(x) = 1 - 2/(exp(2x)+1); exp via exp2, division via rcp.
    float t = __builtin_amdgcn_exp2f(x * 2.8853900817779268f); // 2*log2(e)
    return 1.0f - 2.0f * __builtin_amdgcn_rcpf(t + 1.0f);
}

__device__ __forceinline__ v8f wmma_f16(const HV& A, const HV& B, v8f acc) {
    return __builtin_amdgcn_wmma_f32_16x16x32_f16(
        false, A.v16, false, B.v16, (short)0, acc, false, false);
}

// ---------------------------------------------------------------------------
// Prep kernel: build per-lane-swizzled f16 B-fragments + padded f32 biases in ws.
// B fragment layout (16-bit 32x16 K x N, wave32):
//   lane L: N = t*16 + (L%16); halves h=0..7 -> K = c*32 + (L<16?0:8) + h, h=8..15 -> K = c*32 + (L<16?0:8) + 16 + (h-8)
// Padding: K >= fan_in or N >= fan_out -> 0.
// ---------------------------------------------------------------------------
__global__ void PINN_prep_kernel(
    const float* W0, const float* W1, const float* W2, const float* W3,
    const float* W4, const float* W5, const float* W6,
    const float* b0, const float* b1, const float* b2, const float* b3,
    const float* b4, const float* b5, const float* b6,
    _Float16* __restrict__ wsFrag, float* __restrict__ wsBias)
{
    const float* Ws[7] = {W0, W1, W2, W3, W4, W5, W6};
    const float* Bs[7] = {b0, b1, b2, b3, b4, b5, b6};
    const int fin[7]  = {3, 40, 40, 40, 40, 40, 40};
    const int fout[7] = {40, 40, 40, 40, 40, 40, 3};

    int id = blockIdx.x * blockDim.x + threadIdx.x;
    if (id < FRAG_HALVES) {
        int f = id >> 9;            // fragment index (512 halves each)
        int r = id & 511;
        int lane = r >> 4;
        int h = r & 15;
        int t = f % 3;
        int c = (f / 3) & 1;
        int l = f / 6;
        int K = c * 32 + ((lane & 16) ? 8 : 0) + (h < 8 ? h : h + 8);
        int n = t * 16 + (lane & 15);
        float v = 0.0f;
        if (K < fin[l] && n < fout[l]) v = Ws[l][K * fout[l] + n];
        wsFrag[id] = (_Float16)v;
    } else if (id < FRAG_HALVES + BIAS_FLOATS) {
        int j = id - FRAG_HALVES;
        int l = j / 48, n = j % 48;
        wsBias[j] = (n < fout[l]) ? Bs[l][n] : 0.0f;
    }
}

// ---------------------------------------------------------------------------
// Main kernel: one wave = 32 points per iteration (two 16-row WMMA sub-tiles
// sharing each B fragment -> two independent accumulator chains).
// Activations ping-pong in LDS (f16, 32 rows x 64 cols padded, stride 72).
// A fragment layout (16-bit 16x32 M x K, wave32):
//   lane L: M = L%16; halves 0..7 -> K = c*32 + (L<16?0:8) + 0..7
//                     halves 8..15 -> K = c*32 + (L<16?0:8) + 16..23
// C/D layout: VGPR r, lane L: M = (L<16? r : r+8), N = L%16.
// ---------------------------------------------------------------------------
__global__ __launch_bounds__(256) void PINN_mlp_kernel(
    const float* __restrict__ coords,
    const _Float16* __restrict__ wsFrag,
    const float* __restrict__ wsBias,
    float* __restrict__ out,
    int nPairs, int nPoints)
{
    __shared__ __align__(16) _Float16 sFrag[FRAG_HALVES];
    __shared__ float sBias[BIAS_FLOATS];
    __shared__ __align__(16) _Float16 sAct[ACT_HALVES];

    const int tid = threadIdx.x;

    // Block-start: stage weight fragments + biases into LDS, zero act buffers.
    for (int i = tid; i < FRAG_HALVES / 8; i += 256)
        ((v8h*)sFrag)[i] = ((const v8h*)wsFrag)[i];
    for (int i = tid; i < BIAS_FLOATS; i += 256)
        sBias[i] = wsBias[i];
    {
        v8h z = {};
        for (int i = tid; i < ACT_HALVES / 8; i += 256)
            ((v8h*)sAct)[i] = z;
    }
    __syncthreads();

    const int wave = tid >> 5;
    const int lane = tid & 31;
    const int col  = lane & 15;     // N column within tile; also A-row
    const int hi8  = (lane >> 4) * 8;

    _Float16* actA = sAct + (wave * 2 + 0) * ACT_BUF_HALVES;
    _Float16* actB = sAct + (wave * 2 + 1) * ACT_BUF_HALVES;

    const int waveGlobal = blockIdx.x * 8 + wave;
    const int waveCount  = gridDim.x * 8;

    for (int pair = waveGlobal; pair < nPairs; pair += waveCount) {
        // Zero layer-0 input chunk (rows 0..31, cols 0..31). Cols 32..47 are
        // rewritten by layers 1/3/5 before any read; 48..63 stay zero.
        {
            v8h z = {};
            _Float16* zp = actA + lane * XSTR;
            *(v8h*)(zp +  0) = z;
            *(v8h*)(zp +  8) = z;
            *(v8h*)(zp + 16) = z;
            *(v8h*)(zp + 24) = z;
        }
        __builtin_amdgcn_wave_barrier();

        // Coords: lane L -> point pair*32 + L, row L, cols 0..2 (coalesced).
        {
            int p = pair * 32 + lane;
            if (p < nPoints) {
                actA[lane * XSTR + 0] = (_Float16)coords[p * 3 + 0];
                actA[lane * XSTR + 1] = (_Float16)coords[p * 3 + 1];
                actA[lane * XSTR + 2] = (_Float16)coords[p * 3 + 2];
            }
        }
        __builtin_amdgcn_wave_barrier();

        _Float16* aIn  = actA;
        _Float16* aOut = actB;

        // ---- Layer 0: K padded to 32 (single chunk), Linear + tanh ----
        #pragma unroll
        for (int t = 0; t < 3; ++t) {
            float bv = sBias[t * 16 + col];
            v8f acc0, acc1;
            #pragma unroll
            for (int r = 0; r < 8; ++r) { acc0[r] = bv; acc1[r] = bv; }
            HV Bf, A0, A1;
            const _Float16* bp = sFrag + ((t) * 32 + lane) * 16;
            Bf.v8[0] = *(const v8h*)(bp);
            Bf.v8[1] = *(const v8h*)(bp + 8);
            const _Float16* ap0 = aIn + col * XSTR + hi8;
            A0.v8[0] = *(const v8h*)(ap0);
            A0.v8[1] = *(const v8h*)(ap0 + 16);
            const _Float16* ap1 = ap0 + 16 * XSTR;
            A1.v8[0] = *(const v8h*)(ap1);
            A1.v8[1] = *(const v8h*)(ap1 + 16);
            acc0 = wmma_f16(A0, Bf, acc0);
            acc1 = wmma_f16(A1, Bf, acc1);
            _Float16* op0 = aOut + hi8 * XSTR + t * 16 + col;
            _Float16* op1 = op0 + 16 * XSTR;
            #pragma unroll
            for (int r = 0; r < 8; ++r) {
                op0[r * XSTR] = (_Float16)tanh_fast(acc0[r]);
                op1[r * XSTR] = (_Float16)tanh_fast(acc1[r]);
            }
        }
        __builtin_amdgcn_wave_barrier();
        { _Float16* tmp = aIn; aIn = aOut; aOut = tmp; }

        // ---- Layers 1..5: K = 64 (two chunks), Linear + tanh ----
        for (int l = 1; l < 6; ++l) {
            #pragma unroll
            for (int t = 0; t < 3; ++t) {
                float bv = sBias[l * 48 + t * 16 + col];
                v8f acc0, acc1;
                #pragma unroll
                for (int r = 0; r < 8; ++r) { acc0[r] = bv; acc1[r] = bv; }
                #pragma unroll
                for (int c = 0; c < 2; ++c) {
                    HV Bf, A0, A1;
                    const _Float16* bp =
                        sFrag + (((l * 2 + c) * 3 + t) * 32 + lane) * 16;
                    Bf.v8[0] = *(const v8h*)(bp);
                    Bf.v8[1] = *(const v8h*)(bp + 8);
                    const _Float16* ap0 = aIn + col * XSTR + c * 32 + hi8;
                    A0.v8[0] = *(const v8h*)(ap0);
                    A0.v8[1] = *(const v8h*)(ap0 + 16);
                    const _Float16* ap1 = ap0 + 16 * XSTR;
                    A1.v8[0] = *(const v8h*)(ap1);
                    A1.v8[1] = *(const v8h*)(ap1 + 16);
                    acc0 = wmma_f16(A0, Bf, acc0);
                    acc1 = wmma_f16(A1, Bf, acc1);
                }
                _Float16* op0 = aOut + hi8 * XSTR + t * 16 + col;
                _Float16* op1 = op0 + 16 * XSTR;
                #pragma unroll
                for (int r = 0; r < 8; ++r) {
                    op0[r * XSTR] = (_Float16)tanh_fast(acc0[r]);
                    op1[r * XSTR] = (_Float16)tanh_fast(acc1[r]);
                }
            }
            __builtin_amdgcn_wave_barrier();
            _Float16* tmp = aIn; aIn = aOut; aOut = tmp;
        }

        // ---- Layer 6: Linear only, N-tile 0 (cols 0..2 valid) ----
        {
            float bv = sBias[6 * 48 + col];
            v8f acc0, acc1;
            #pragma unroll
            for (int r = 0; r < 8; ++r) { acc0[r] = bv; acc1[r] = bv; }
            #pragma unroll
            for (int c = 0; c < 2; ++c) {
                HV Bf, A0, A1;
                const _Float16* bp = sFrag + (((12 + c) * 3) * 32 + lane) * 16;
                Bf.v8[0] = *(const v8h*)(bp);
                Bf.v8[1] = *(const v8h*)(bp + 8);
                const _Float16* ap0 = aIn + col * XSTR + c * 32 + hi8;
                A0.v8[0] = *(const v8h*)(ap0);
                A0.v8[1] = *(const v8h*)(ap0 + 16);
                const _Float16* ap1 = ap0 + 16 * XSTR;
                A1.v8[0] = *(const v8h*)(ap1);
                A1.v8[1] = *(const v8h*)(ap1 + 16);
                acc0 = wmma_f16(A0, Bf, acc0);
                acc1 = wmma_f16(A1, Bf, acc1);
            }
            if (col < 3) {
                #pragma unroll
                for (int r = 0; r < 8; ++r) {
                    int p0 = pair * 32 + hi8 + r;
                    if (p0 < nPoints) out[p0 * 3 + col] = acc0[r];
                    int p1 = p0 + 16;
                    if (p1 < nPoints) out[p1 * 3 + col] = acc1[r];
                }
            }
        }
        __builtin_amdgcn_wave_barrier();
    }
}

extern "C" void kernel_launch(void* const* d_in, const int* in_sizes, int n_in,
                              void* d_out, int out_size, void* d_ws, size_t ws_size,
                              hipStream_t stream) {
    const float* coords = (const float*)d_in[0];
    const float* W[7];
    const float* B[7];
    for (int i = 0; i < 7; ++i) {
        W[i] = (const float*)d_in[1 + 2 * i];
        B[i] = (const float*)d_in[2 + 2 * i];
    }
    _Float16* wsFrag = (_Float16*)d_ws;
    float* wsBias = (float*)((char*)d_ws + FRAG_HALVES * sizeof(_Float16));

    int nPoints = in_sizes[0] / 3;
    int nPairs = (nPoints + 31) / 32;   // 32 points per wave-iteration

    // Build swizzled weight fragments + padded biases in workspace.
    int prepThreads = FRAG_HALVES + BIAS_FLOATS;
    int prepBlocks = (prepThreads + 255) / 256;
    PINN_prep_kernel<<<prepBlocks, 256, 0, stream>>>(
        W[0], W[1], W[2], W[3], W[4], W[5], W[6],
        B[0], B[1], B[2], B[3], B[4], B[5], B[6],
        wsFrag, wsBias);

    int blocks = 2048;
    int maxBlocks = (nPairs + 7) / 8;
    if (blocks > maxBlocks) blocks = maxBlocks;
    if (blocks < 1) blocks = 1;
    PINN_mlp_kernel<<<blocks, 256, 0, stream>>>(
        coords, wsFrag, wsBias, (float*)d_out, nPairs, nPoints);
}
